// OURS_3135326126814
// MI455X (gfx1250) — compile-verified, matrix operand order checked by reading, not run on hardware
//
#include <hip/hip_runtime.h>
#include <hip/hip_bf16.h>

typedef _Float16 h16;
typedef __attribute__((ext_vector_type(16))) _Float16 v16h;
typedef __attribute__((ext_vector_type(8)))  _Float16 v8h;
typedef __attribute__((ext_vector_type(8)))  float    v8f;
typedef __attribute__((ext_vector_type(4)))  float    v4f;

// ---------------------------------------------------------------------------
// Fragment loaders
// ---------------------------------------------------------------------------
template<int AF16>
__device__ __forceinline__ v16h load_afrag(const void* __restrict__ A, size_t off)
{
  v16h a;
  if (AF16) {
    const h16* p = (const h16*)A + off;
    v8h lo = *(const v8h*)p;
    v8h hi = *(const v8h*)(p + 16);
#pragma unroll
    for (int i = 0; i < 8; i++) { a[i] = lo[i]; a[8 + i] = hi[i]; }
  } else {
    const float* p = (const float*)A + off;
    v4f f0 = *(const v4f*)p;
    v4f f1 = *(const v4f*)(p + 4);
    v4f f2 = *(const v4f*)(p + 16);
    v4f f3 = *(const v4f*)(p + 20);
#pragma unroll
    for (int i = 0; i < 4; i++) {
      a[i]      = (h16)f0[i]; a[4 + i]  = (h16)f1[i];
      a[8 + i]  = (h16)f2[i]; a[12 + i] = (h16)f3[i];
    }
  }
  return a;
}

__device__ __forceinline__ v16h load_bfrag(const h16* __restrict__ p)
{
  v8h lo = *(const v8h*)p;
  v8h hi = *(const v8h*)(p + 16);
  v16h b;
#pragma unroll
  for (int i = 0; i < 8; i++) { b[i] = lo[i]; b[8 + i] = hi[i]; }
  return b;
}

// ---------------------------------------------------------------------------
// Generic batched WMMA GEMM:  C[z] = act( A[z] (MxK) * B^T[z] + bias + Res )
// B is stored TRANSPOSED: f16 [N][K] rows (row stride ldb) -> contiguous frags.
// Block: 128 threads = 4 waves; each wave computes a 16x64 strip of a 64x64
// block tile using v_wmma_f32_16x16x32_f16, with a software-pipelined k-loop
// (next-step fragments prefetched into distinct registers before the WMMAs).
// ---------------------------------------------------------------------------
struct GemmArgs {
  const void* A; const h16* B; const float* bias; const float* Res; void* C;
  int K, lda, ldb, ldc;
  long aOut; int aInN; long aIn;   // batch offset = (z/aInN)*aOut + (z%aInN)*aIn
  long bOut; int bInN; long bIn;
  long cOut; int cInN; long cIn;   // Res uses same offsets/ldc as C
};

template<int AF16, int SMODE, int RELU, int RES>
__global__ __launch_bounds__(128) void k_gemm(GemmArgs g)
{
  const int wid  = threadIdx.x >> 5;
  const int lane = threadIdx.x & 31;
  const int ln15 = lane & 15;
  const int z    = blockIdx.z;
  const size_t aoff = (size_t)(z / g.aInN) * g.aOut + (size_t)(z % g.aInN) * g.aIn;
  const size_t boff = (size_t)(z / g.bInN) * g.bOut + (size_t)(z % g.bInN) * g.bIn;
  const size_t coff = (size_t)(z / g.cInN) * g.cOut + (size_t)(z % g.cInN) * g.cIn;
  const int m0  = blockIdx.x * 64 + wid * 16;
  const int n0  = blockIdx.y * 64;
  const int mr  = m0 + ln15;
  const int kph = (lane >> 4) << 3;        // 0 for lanes 0-15, 8 for lanes 16-31

  // loop-invariant fragment base addresses
  const size_t arow = aoff + (size_t)mr * g.lda + kph;
  const h16* bp0 = g.B + boff + (size_t)(n0 +  0 + ln15) * g.ldb + kph;
  const h16* bp1 = g.B + boff + (size_t)(n0 + 16 + ln15) * g.ldb + kph;
  const h16* bp2 = g.B + boff + (size_t)(n0 + 32 + ln15) * g.ldb + kph;
  const h16* bp3 = g.B + boff + (size_t)(n0 + 48 + ln15) * g.ldb + kph;

  v8f acc0 = {}, acc1 = {}, acc2 = {}, acc3 = {};

  // prologue: load k-step 0 fragments
  v16h aC  = load_afrag<AF16>(g.A, arow);
  v16h bC0 = load_bfrag(bp0);
  v16h bC1 = load_bfrag(bp1);
  v16h bC2 = load_bfrag(bp2);
  v16h bC3 = load_bfrag(bp3);

  int k0 = 0;
  for (; k0 + 32 < g.K; k0 += 32) {
    const int kn = k0 + 32;
    // prefetch next k-step into fresh registers
    v16h aN  = load_afrag<AF16>(g.A, arow + kn);
    v16h bN0 = load_bfrag(bp0 + kn);
    v16h bN1 = load_bfrag(bp1 + kn);
    v16h bN2 = load_bfrag(bp2 + kn);
    v16h bN3 = load_bfrag(bp3 + kn);
    // compute with current fragments while next loads are in flight
    acc0 = __builtin_amdgcn_wmma_f32_16x16x32_f16(false, aC, false, bC0, (short)0, acc0, false, false);
    acc1 = __builtin_amdgcn_wmma_f32_16x16x32_f16(false, aC, false, bC1, (short)0, acc1, false, false);
    acc2 = __builtin_amdgcn_wmma_f32_16x16x32_f16(false, aC, false, bC2, (short)0, acc2, false, false);
    acc3 = __builtin_amdgcn_wmma_f32_16x16x32_f16(false, aC, false, bC3, (short)0, acc3, false, false);
    aC = aN; bC0 = bN0; bC1 = bN1; bC2 = bN2; bC3 = bN3;
  }
  // epilogue k-step
  acc0 = __builtin_amdgcn_wmma_f32_16x16x32_f16(false, aC, false, bC0, (short)0, acc0, false, false);
  acc1 = __builtin_amdgcn_wmma_f32_16x16x32_f16(false, aC, false, bC1, (short)0, acc1, false, false);
  acc2 = __builtin_amdgcn_wmma_f32_16x16x32_f16(false, aC, false, bC2, (short)0, acc2, false, false);
  acc3 = __builtin_amdgcn_wmma_f32_16x16x32_f16(false, aC, false, bC3, (short)0, acc3, false, false);

  const int mbase = m0 + kph;              // kph == 0/8 == row-half offset for C
  v8f accs[4] = {acc0, acc1, acc2, acc3};
#pragma unroll
  for (int t = 0; t < 4; t++) {
    const int n = n0 + t * 16 + ln15;
    const float bv = g.bias ? g.bias[n] : 0.0f;
#pragma unroll
    for (int v = 0; v < 8; v++) {
      const int m = mbase + v;
      float c = accs[t][v] + bv;
      if (RES)  c += g.Res[coff + (size_t)m * g.ldc + n];
      if (RELU) c = fmaxf(c, 0.0f);
      if (SMODE == 0)      ((float*)g.C)[coff + (size_t)m * g.ldc + n] = c;
      else if (SMODE == 1) ((h16*) g.C)[coff + (size_t)m * g.ldc + n] = (h16)c;
      else                 ((h16*) g.C)[coff + (size_t)n * g.ldc + m] = (h16)c; // transposed
    }
  }
}

// ---------------------------------------------------------------------------
// Elementwise / reduction helper kernels
// ---------------------------------------------------------------------------
__global__ __launch_bounds__(256) void k_wtconv(const float* __restrict__ src,
                                                h16* __restrict__ dst, int Kd, int Nd)
{
  int idx = blockIdx.x * 256 + threadIdx.x;
  if (idx < Kd * Nd) {
    int k = idx / Nd, n = idx % Nd;
    dst[(size_t)n * Kd + k] = (h16)src[idx];
  }
}

__global__ __launch_bounds__(256) void k_bnstats(const float* __restrict__ x,
    const float* __restrict__ g, const float* __restrict__ b,
    float* __restrict__ scale, float* __restrict__ shift)
{
  __shared__ float r1[256], r2[256];
  const int f = blockIdx.x;
  float s = 0.f, s2 = 0.f;
  for (int i = threadIdx.x; i < 16 * 512; i += 256) {
    float v = x[(size_t)i * 64 + f];
    s += v; s2 += v * v;
  }
  r1[threadIdx.x] = s; r2[threadIdx.x] = s2; __syncthreads();
  for (int k = 128; k > 0; k >>= 1) {
    if (threadIdx.x < k) { r1[threadIdx.x] += r1[threadIdx.x + k]; r2[threadIdx.x] += r2[threadIdx.x + k]; }
    __syncthreads();
  }
  if (threadIdx.x == 0) {
    float m   = r1[0] * (1.f / 8192.f);
    float var = r2[0] * (1.f / 8192.f) - m * m;
    float sc  = g[f] * rsqrtf(var + 1e-5f);
    scale[f] = sc; shift[f] = b[f] - m * sc;
  }
}

__global__ __launch_bounds__(128) void k_msg(const float* __restrict__ enc,
                                             const int* __restrict__ tgt,
                                             float* __restrict__ msg)
{
  __shared__ float r[128];
  const int b = blockIdx.x;
  const int t0 = tgt[0];
  float s = 0.f;
  for (int t = threadIdx.x; t < 512; t += 128)
    s += enc[((size_t)b * 512 + t) * 64 + t0];
  r[threadIdx.x] = s; __syncthreads();
  for (int k = 64; k > 0; k >>= 1) {
    if (threadIdx.x < k) r[threadIdx.x] += r[threadIdx.x + k];
    __syncthreads();
  }
  if (threadIdx.x == 0) msg[b] = r[0] * (1.f / 512.f);
}

__global__ __launch_bounds__(256) void k_bnapply(const float* __restrict__ x,
    const float* __restrict__ scale, const float* __restrict__ shift,
    float* __restrict__ xn, float* __restrict__ xnT)
{
  int idx = blockIdx.x * 256 + threadIdx.x;  // < 16*512*64
  int f = idx & 63, t = (idx >> 6) & 511, b = idx >> 15;
  float v = x[idx] * scale[f] + shift[f];
  xn[idx] = v;
  xnT[((size_t)b * 64 + f) * 512 + t] = v;
}

__device__ __forceinline__ float posenc_val(int pos, int d)
{
  int j = d >> 1;
  float ang = (float)pos * __expf((float)(2 * j) * (-9.210340371976184f / 512.f));
  return (d & 1) ? __cosf(ang) : __sinf(ang);
}

__global__ __launch_bounds__(256) void k_timeembed(float* __restrict__ xt,
    const int* __restrict__ stamp,
    const float* __restrict__ Eh, const float* __restrict__ Ew,
    const float* __restrict__ Ed, const float* __restrict__ Em)
{
  int idx = blockIdx.x * 256 + threadIdx.x;  // < 16*512*512
  int d = idx & 511, t = (idx >> 9) & 511, b = idx >> 18;
  const int* sp = stamp + ((size_t)b * 512 + t) * 4;
  xt[idx] += posenc_val(t, d)
           + Eh[(size_t)sp[3] * 512 + d] + Ew[(size_t)sp[2] * 512 + d]
           + Ed[(size_t)sp[1] * 512 + d] + Em[(size_t)sp[0] * 512 + d];
}

__global__ __launch_bounds__(256) void k_featpe(float* __restrict__ xf)
{
  int idx = blockIdx.x * 256 + threadIdx.x;  // < 16*64*512
  int d = idx & 511, f = (idx >> 9) & 63;
  xf[idx] += posenc_val(f, d);
}

__global__ __launch_bounds__(128) void k_ln(const float* __restrict__ in,
    const float* __restrict__ g, const float* __restrict__ b,
    float* __restrict__ out)
{
  __shared__ float r[128];
  const size_t base = (size_t)blockIdx.x * 512;
  float v[4];
  float s = 0.f;
#pragma unroll
  for (int i = 0; i < 4; i++) { v[i] = in[base + threadIdx.x + i * 128]; s += v[i]; }
  r[threadIdx.x] = s; __syncthreads();
  for (int k = 64; k > 0; k >>= 1) {
    if (threadIdx.x < k) r[threadIdx.x] += r[threadIdx.x + k];
    __syncthreads();
  }
  float mean = r[0] * (1.f / 512.f); __syncthreads();
  s = 0.f;
#pragma unroll
  for (int i = 0; i < 4; i++) { float d = v[i] - mean; s += d * d; }
  r[threadIdx.x] = s; __syncthreads();
  for (int k = 64; k > 0; k >>= 1) {
    if (threadIdx.x < k) r[threadIdx.x] += r[threadIdx.x + k];
    __syncthreads();
  }
  float istd = rsqrtf(r[0] * (1.f / 512.f) + 1e-5f);
#pragma unroll
  for (int i = 0; i < 4; i++) {
    int c = threadIdx.x + i * 128;
    out[base + c] = (v[i] - mean) * istd * g[c] + b[c];
  }
}

// causal softmax over row of length Lk (cols j<=q allowed); writes f16 probs
__global__ __launch_bounds__(128) void k_softmax(const float* __restrict__ S,
                                                 h16* __restrict__ P, int Lk)
{
  __shared__ float r[128];
  const int q = blockIdx.x;
  const size_t base = ((size_t)blockIdx.y * gridDim.x + q) * (size_t)Lk;
  const int nall = min(q + 1, Lk);
  float mx = -3.0e38f;
  for (int j = threadIdx.x; j < nall; j += 128) mx = fmaxf(mx, S[base + j]);
  r[threadIdx.x] = mx; __syncthreads();
  for (int k = 64; k > 0; k >>= 1) {
    if (threadIdx.x < k) r[threadIdx.x] = fmaxf(r[threadIdx.x], r[threadIdx.x + k]);
    __syncthreads();
  }
  mx = r[0]; __syncthreads();
  float s = 0.f;
  for (int j = threadIdx.x; j < nall; j += 128) s += __expf(S[base + j] - mx);
  r[threadIdx.x] = s; __syncthreads();
  for (int k = 64; k > 0; k >>= 1) {
    if (threadIdx.x < k) r[threadIdx.x] += r[threadIdx.x + k];
    __syncthreads();
  }
  const float inv = 1.f / r[0];
  for (int j = threadIdx.x; j < Lk; j += 128)
    P[base + j] = (h16)((j < nall) ? __expf(S[base + j] - mx) * inv : 0.f);
}

__global__ __launch_bounds__(256) void k_head(const float* __restrict__ X,
    const float* __restrict__ W, const float* __restrict__ bias,
    float* __restrict__ out, int Kd)
{
  __shared__ float r[256];
  const int p = blockIdx.x, b = blockIdx.y;
  const float* x = X + (size_t)b * Kd;
  float s = 0.f;
  for (int i = threadIdx.x; i < Kd; i += 256) s += x[i] * W[(size_t)i * 32 + p];
  r[threadIdx.x] = s; __syncthreads();
  for (int k = 128; k > 0; k >>= 1) {
    if (threadIdx.x < k) r[threadIdx.x] += r[threadIdx.x + k];
    __syncthreads();
  }
  if (threadIdx.x == 0) out[(size_t)b * 32 + p] = r[0] + bias[p];
}

__global__ void k_mix(const float* __restrict__ ot, const float* __restrict__ otf,
                      const float* __restrict__ msg, const float* __restrict__ w,
                      float* __restrict__ out)
{
  int i = blockIdx.x * 64 + threadIdx.x;
  if (i < 512)
    out[i] = otf[i] * w[0] + ot[i] * w[1] + w[3] + msg[i >> 5] * w[2];
}

// ---------------------------------------------------------------------------
// Host-side orchestration
// ---------------------------------------------------------------------------
struct LayerW {
  const float *bq, *bk, *bv, *bo, *ln1g, *ln1b, *ln2g, *ln2b, *b1, *b2;
  const h16 *wqT, *wkT, *wvT, *woT, *w1T, *w2T;
};
struct Scr {
  h16 *Qh, *Kh, *VTh, *Ph, *FF1h;
  float *S, *O, *X1pre, *X1, *X2pre;
};

static void gemm_launch(hipStream_t st, int af16, int smode, int relu, int res,
    const void* A, const h16* B, const float* bias, const float* Res, void* C,
    int M, int N, int K, int lda, int ldb, int ldc,
    long aO, int aN, long aI, long bO, int bN, long bI, long cO, int cN, long cI,
    int nb)
{
  GemmArgs g{A, B, bias, Res, C, K, lda, ldb, ldc, aO, aN, aI, bO, bN, bI, cO, cN, cI};
  dim3 gr(M / 64, N / 64, nb), bl(128, 1, 1);
  if (!af16 && smode == 0 && !res)        k_gemm<0,0,0,0><<<gr, bl, 0, st>>>(g);
  else if (!af16 && smode == 0 && res)    k_gemm<0,0,0,1><<<gr, bl, 0, st>>>(g);
  else if (!af16 && smode == 1 && !relu)  k_gemm<0,1,0,0><<<gr, bl, 0, st>>>(g);
  else if (!af16 && smode == 1 && relu)   k_gemm<0,1,1,0><<<gr, bl, 0, st>>>(g);
  else if (!af16 && smode == 2)           k_gemm<0,2,0,0><<<gr, bl, 0, st>>>(g);
  else if ( af16 && smode == 0 && !res)   k_gemm<1,0,0,0><<<gr, bl, 0, st>>>(g);
  else                                    k_gemm<1,0,0,1><<<gr, bl, 0, st>>>(g);
}

static void enc_layer(hipStream_t st, const Scr& sc, const LayerW& P,
                      const float* xq, int Lq, const float* xkv, int Lk, float* out)
{
  // Q, K projections (f16 out), V projection stored transposed per head
  gemm_launch(st, 0, 1, 0, 0, xq,  P.wqT, P.bq, nullptr, sc.Qh,  Lq, 512, 512, 512, 512, 512,
              (long)Lq * 512, 1, 0, 0, 1, 0, (long)Lq * 512, 1, 0, 16);
  gemm_launch(st, 0, 1, 0, 0, xkv, P.wkT, P.bk, nullptr, sc.Kh,  Lk, 512, 512, 512, 512, 512,
              (long)Lk * 512, 1, 0, 0, 1, 0, (long)Lk * 512, 1, 0, 16);
  gemm_launch(st, 0, 2, 0, 0, xkv, P.wvT, P.bv, nullptr, sc.VTh, Lk, 512, 512, 512, 512, Lk,
              (long)Lk * 512, 1, 0, 0, 1, 0, (long)512 * Lk, 1, 0, 16);
  // S = Q K^T   (batch = B*H = 128)
  gemm_launch(st, 1, 0, 0, 0, sc.Qh, sc.Kh, nullptr, nullptr, sc.S, Lq, Lk, 64, 512, 512, Lk,
              (long)Lq * 512, 8, 64, (long)Lk * 512, 8, 64, (long)Lq * Lk, 1, 0, 128);
  // causal softmax -> f16 probs
  k_softmax<<<dim3(Lq, 128), 128, 0, st>>>(sc.S, sc.Ph, Lk);
  // O = P V   (batch = 128), heads re-interleaved into [B, Lq, 512]
  gemm_launch(st, 1, 0, 0, 0, sc.Ph, sc.VTh, nullptr, nullptr, sc.O, Lq, 64, Lk, Lk, Lk, 512,
              (long)Lq * Lk, 1, 0, (long)64 * Lk, 1, 0, (long)Lq * 512, 8, 64, 128);
  // attn = O wo + bo + xq (residual), then LN1
  gemm_launch(st, 0, 0, 0, 1, sc.O, P.woT, P.bo, xq, sc.X1pre, Lq, 512, 512, 512, 512, 512,
              (long)Lq * 512, 1, 0, 0, 1, 0, (long)Lq * 512, 1, 0, 16);
  k_ln<<<16 * Lq, 128, 0, st>>>(sc.X1pre, P.ln1g, P.ln1b, sc.X1);
  // FFN: relu(X1 w1 + b1) w2 + b2 + X1, then LN2
  gemm_launch(st, 0, 1, 1, 0, sc.X1, P.w1T, P.b1, nullptr, sc.FF1h, Lq, 2048, 512, 512, 512, 2048,
              (long)Lq * 512, 1, 0, 0, 1, 0, (long)Lq * 2048, 1, 0, 16);
  gemm_launch(st, 1, 0, 0, 1, sc.FF1h, P.w2T, P.b2, sc.X1, sc.X2pre, Lq, 512, 2048, 2048, 2048, 512,
              (long)Lq * 2048, 1, 0, 0, 1, 0, (long)Lq * 512, 1, 0, 16);
  k_ln<<<16 * Lq, 128, 0, st>>>(sc.X2pre, P.ln2g, P.ln2b, out);
}

extern "C" void kernel_launch(void* const* d_in, const int* in_sizes, int n_in,
                              void* d_out, int out_size, void* d_ws, size_t ws_size,
                              hipStream_t stream)
{
  (void)in_sizes; (void)out_size; (void)ws_size;
  if (n_in < 84) return;
  auto F32 = [&](int i) { return (const float*)d_in[i]; };
  auto I32 = [&](int i) { return (const int*)d_in[i]; };

  const float* enc_in   = F32(0);
  const float* dec_in   = F32(1);
  const float* bn_g     = F32(2);
  const float* bn_b     = F32(3);
  const float* tok_t_w  = F32(4);
  const float* tok_t_b  = F32(5);
  const float* tok_f_w  = F32(6);
  const float* tok_f_b  = F32(7);
  const float* emb_hour = F32(8);
  const float* emb_wday = F32(9);
  const float* emb_day  = F32(10);
  const float* emb_mon  = F32(11);
  const int ENC_TIME = 12, ENC_FEAT = 28, ENC_TF = 44, ENC_FT = 60;
  const float* out_t_w  = F32(76);
  const float* out_t_b  = F32(77);
  const float* out_tf_w = F32(78);
  const float* out_tf_b = F32(79);
  const float* mix_w    = F32(80);
  const int*   dec_stamp   = I32(82);
  const int*   pred_target = I32(83);

  size_t off = 0;
  auto alloc = [&](size_t bytes) -> void* {
    void* p = (char*)d_ws + off;
    off = (off + bytes + 255) & ~(size_t)255;
    return p;
  };

  // ---- f16 transposed weights ----
  h16* tokTT = (h16*)alloc((size_t)512 * 64 * 2);
  h16* tokFT = (h16*)alloc((size_t)512 * 512 * 2);
  k_wtconv<<<(64 * 512 + 255) / 256, 256, 0, stream>>>(tok_t_w, tokTT, 64, 512);
  k_wtconv<<<(512 * 512 + 255) / 256, 256, 0, stream>>>(tok_f_w, tokFT, 512, 512);

  struct Inst { int base, i; };
  const Inst insts[6] = {{ENC_TIME,0},{ENC_TIME,1},{ENC_FEAT,0},{ENC_FEAT,1},{ENC_TF,0},{ENC_FT,0}};
  LayerW LW[6];
  for (int li = 0; li < 6; li++) {
    const int bs = insts[li].base, i = insts[li].i;
    const float* wq = F32(bs + 0)  + (size_t)i * 512 * 512;
    const float* wk = F32(bs + 2)  + (size_t)i * 512 * 512;
    const float* wv = F32(bs + 4)  + (size_t)i * 512 * 512;
    const float* wo = F32(bs + 6)  + (size_t)i * 512 * 512;
    const float* w1 = F32(bs + 12) + (size_t)i * 512 * 2048;
    const float* w2 = F32(bs + 14) + (size_t)i * 2048 * 512;
    h16* wqT = (h16*)alloc((size_t)512 * 512 * 2);
    h16* wkT = (h16*)alloc((size_t)512 * 512 * 2);
    h16* wvT = (h16*)alloc((size_t)512 * 512 * 2);
    h16* woT = (h16*)alloc((size_t)512 * 512 * 2);
    h16* w1T = (h16*)alloc((size_t)512 * 2048 * 2);
    h16* w2T = (h16*)alloc((size_t)2048 * 512 * 2);
    k_wtconv<<<(512 * 512 + 255) / 256, 256, 0, stream>>>(wq, wqT, 512, 512);
    k_wtconv<<<(512 * 512 + 255) / 256, 256, 0, stream>>>(wk, wkT, 512, 512);
    k_wtconv<<<(512 * 512 + 255) / 256, 256, 0, stream>>>(wv, wvT, 512, 512);
    k_wtconv<<<(512 * 512 + 255) / 256, 256, 0, stream>>>(wo, woT, 512, 512);
    k_wtconv<<<(512 * 2048 + 255) / 256, 256, 0, stream>>>(w1, w1T, 512, 2048);
    k_wtconv<<<(2048 * 512 + 255) / 256, 256, 0, stream>>>(w2, w2T, 2048, 512);
    LW[li] = { F32(bs + 1)  + (size_t)i * 512,  F32(bs + 3)  + (size_t)i * 512,
               F32(bs + 5)  + (size_t)i * 512,  F32(bs + 7)  + (size_t)i * 512,
               F32(bs + 8)  + (size_t)i * 512,  F32(bs + 9)  + (size_t)i * 512,
               F32(bs + 10) + (size_t)i * 512,  F32(bs + 11) + (size_t)i * 512,
               F32(bs + 13) + (size_t)i * 2048, F32(bs + 15) + (size_t)i * 512,
               wqT, wkT, wvT, woT, w1T, w2T };
  }

  // ---- activation buffers ----
  float* bscale = (float*)alloc(64 * 4);
  float* bshift = (float*)alloc(64 * 4);
  float* msg    = (float*)alloc(16 * 4);
  float* outT   = (float*)alloc(512 * 4);
  float* outTF  = (float*)alloc(512 * 4);
  float* xn  = (float*)alloc((size_t)524288 * 4);
  float* xnT = (float*)alloc((size_t)524288 * 4);
  float* xt  = (float*)alloc((size_t)4194304 * 4);
  float* xf  = (float*)alloc((size_t)524288 * 4);
  float* xtf = (float*)alloc((size_t)4194304 * 4);
  float* xft = (float*)alloc((size_t)524288 * 4);

  Scr sc;
  sc.Qh  = (h16*)alloc((size_t)4194304 * 2);
  sc.Kh  = (h16*)alloc((size_t)4194304 * 2);
  sc.VTh = (h16*)alloc((size_t)4194304 * 2);
  sc.Ph  = (h16*)alloc((size_t)33554432 * 2);
  char* Sreg = (char*)alloc((size_t)33554432 * 4);   // 134 MB score region
  sc.S     = (float*)Sreg;                            // live only QK^T -> softmax
  sc.O     = (float*)(Sreg + ((size_t)0  << 20));     // reuses dead S region
  sc.X1pre = (float*)(Sreg + ((size_t)16 << 20));
  sc.X1    = (float*)(Sreg + ((size_t)32 << 20));
  sc.X2pre = (float*)(Sreg + ((size_t)48 << 20));
  sc.FF1h  = (h16*) (Sreg + ((size_t)64 << 20));

  // ---- preprocessing ----
  k_bnstats<<<64, 256, 0, stream>>>(dec_in, bn_g, bn_b, bscale, bshift);
  k_msg<<<16, 128, 0, stream>>>(enc_in, pred_target, msg);
  k_bnapply<<<2048, 256, 0, stream>>>(dec_in, bscale, bshift, xn, xnT);

  // xt = xn @ tok_t_w + b, then +PE +temporal embeddings
  gemm_launch(stream, 0, 0, 0, 0, xn, tokTT, tok_t_b, nullptr, xt, 512, 512, 64, 64, 64, 512,
              (long)512 * 64, 1, 0, 0, 1, 0, (long)512 * 512, 1, 0, 16);
  k_timeembed<<<16384, 256, 0, stream>>>(xt, dec_stamp, emb_hour, emb_wday, emb_day, emb_mon);

  // xf = xn^T @ tok_f_w + b, then +PE
  gemm_launch(stream, 0, 0, 0, 0, xnT, tokFT, tok_f_b, nullptr, xf, 64, 512, 512, 512, 512, 512,
              (long)64 * 512, 1, 0, 0, 1, 0, (long)64 * 512, 1, 0, 16);
  k_featpe<<<2048, 256, 0, stream>>>(xf);

  // ---- encoder stack ----
  enc_layer(stream, sc, LW[0], xt, 512, xt, 512, xt);
  enc_layer(stream, sc, LW[1], xt, 512, xt, 512, xt);
  enc_layer(stream, sc, LW[2], xf, 64,  xf, 64,  xf);
  enc_layer(stream, sc, LW[3], xf, 64,  xf, 64,  xf);
  enc_layer(stream, sc, LW[4], xt, 512, xf, 64,  xtf);
  enc_layer(stream, sc, LW[5], xf, 64,  xt, 512, xft);

  // ---- output heads + mix ----
  k_head<<<dim3(32, 16), 256, 0, stream>>>(xtf, out_t_w,  out_t_b,  outT,  512 * 512);
  k_head<<<dim3(32, 16), 256, 0, stream>>>(xft, out_tf_w, out_tf_b, outTF, 64 * 512);
  k_mix<<<8, 64, 0, stream>>>(outT, outTF, msg, mix_w, (float*)d_out);
}
